// GAT_64037962383824
// MI455X (gfx1250) — matrix operand order repaired and checked by reading, not compiled, not run on hardware
//
#include <hip/hip_runtime.h>

// ---------------------------------------------------------------------------
// Problem constants (from reference)
// ---------------------------------------------------------------------------
static constexpr int NN   = 50000;    // nodes
static constexpr int EE   = 1600000;  // edges

// ---------------------------------------------------------------------------
// WMMA types (gfx1250, wave32)
// ---------------------------------------------------------------------------
typedef __attribute__((ext_vector_type(16))) __bf16 v16bf;
typedef __attribute__((ext_vector_type(8)))  __bf16 v8bf;
typedef __attribute__((ext_vector_type(8)))  float  v8f;

// A-fragment (16x32 bf16, MxK). Lane holds M = lane&15; in this code element e
// maps to K = half*16 + e  ->  one contiguous 16-float span.
__device__ __forceinline__ v16bf load_a_frag(const float* __restrict__ A,
                                             int row0, int lda, int kbase, int lane) {
  const int m = lane & 15, half = lane >> 4;
  const float* arow = A + (size_t)(row0 + m) * lda + kbase + half * 16;
  const float4 x0 = *reinterpret_cast<const float4*>(arow);
  const float4 x1 = *reinterpret_cast<const float4*>(arow + 4);
  const float4 x2 = *reinterpret_cast<const float4*>(arow + 8);
  const float4 x3 = *reinterpret_cast<const float4*>(arow + 12);
  v16bf a;
  a[0]  = (__bf16)x0.x; a[1]  = (__bf16)x0.y; a[2]  = (__bf16)x0.z; a[3]  = (__bf16)x0.w;
  a[4]  = (__bf16)x1.x; a[5]  = (__bf16)x1.y; a[6]  = (__bf16)x1.z; a[7]  = (__bf16)x1.w;
  a[8]  = (__bf16)x2.x; a[9]  = (__bf16)x2.y; a[10] = (__bf16)x2.z; a[11] = (__bf16)x2.w;
  a[12] = (__bf16)x3.x; a[13] = (__bf16)x3.y; a[14] = (__bf16)x3.z; a[15] = (__bf16)x3.w;
  return a;
}

// ---------------------------------------------------------------------------
// WMMA GEMM: Y[M,NCOL] = A[M,KDIM] @ B[KDIM,NCOL] (+ bias)
// B is staged once per block into LDS as bf16, transposed to [NCOL][KDIM] so a
// B-fragment (lane = column n, K = kbase + half*16 + e) is 16 contiguous bf16
// -> two ds_load_b128 per fragment.
// blockDim = 256 (8 waves) -> 128 rows / block.
// ---------------------------------------------------------------------------
template <int KDIM, int NCOL, bool BIAS>
__global__ void gemm_wmma(const float* __restrict__ A, const float* __restrict__ B,
                          const float* __restrict__ bias, float* __restrict__ Y, int M) {
  __shared__ __bf16 Bt[NCOL * KDIM];

  // stage + transpose + convert B (coalesced global read)
  for (int idx = threadIdx.x; idx < KDIM * NCOL; idx += blockDim.x) {
    const int k = idx / NCOL, n = idx % NCOL;   // NCOL is a power of 2
    Bt[n * KDIM + k] = (__bf16)B[idx];
  }
  __syncthreads();

  const int lane = threadIdx.x & 31;
  const int wave = threadIdx.x >> 5;
  const int row0 = (blockIdx.x * (blockDim.x >> 5) + wave) * 16;
  if (row0 >= M) return;

  const int half = lane >> 4, nlo = lane & 15;
  constexpr int NT = NCOL / 16;
  v8f acc[NT] = {};

#pragma unroll
  for (int kt = 0; kt < KDIM / 32; ++kt) {
    const v16bf a = load_a_frag(A, row0, KDIM, kt * 32, lane);
#pragma unroll
    for (int t = 0; t < NT; ++t) {
      const int n = t * 16 + nlo;
      const v8bf* bp = reinterpret_cast<const v8bf*>(Bt + n * KDIM + kt * 32 + half * 16);
      const v8bf blo = bp[0], bhi = bp[1];
      const v16bf b = __builtin_shufflevector(blo, bhi,
          0, 1, 2, 3, 4, 5, 6, 7, 8, 9, 10, 11, 12, 13, 14, 15);
      acc[t] = __builtin_amdgcn_wmma_f32_16x16x32_bf16(
          /*neg_a=*/false, a, /*neg_b=*/false, b,
          /*c_mod=*/(short)0, acc[t], /*reuse_a=*/false, /*reuse_b=*/false);
    }
  }

#pragma unroll
  for (int t = 0; t < NT; ++t) {
#pragma unroll
    for (int r = 0; r < 8; ++r) {
      const int row = row0 + half * 8 + r;
      const int col = t * 16 + nlo;
      float v = acc[t][r];
      if (BIAS) v += bias[col];
      Y[(size_t)row * NCOL + col] = v;
    }
  }
}

// ---------------------------------------------------------------------------
// BatchNorm stats (C = 64): coalesced grid-stride pass; channel of each thread
// is loop-invariant (stride is a multiple of 64). Block folds 256 -> 64 lanes,
// then per-channel atomic accumulate; tiny kernel finalizes mean / rsqrt(var).
// ---------------------------------------------------------------------------
__global__ void bn_partial(const float* __restrict__ X, int total,
                           float* __restrict__ sums, float* __restrict__ sqs) {
  __shared__ float s1[256], s2[256];
  float sum = 0.f, sq = 0.f;
  for (int idx = blockIdx.x * blockDim.x + threadIdx.x; idx < total;
       idx += gridDim.x * blockDim.x) {
    const float v = X[idx];
    sum += v; sq += v * v;
  }
  s1[threadIdx.x] = sum; s2[threadIdx.x] = sq;
  __syncthreads();
  if (threadIdx.x < 64) {
    const int c = threadIdx.x;  // == channel (tid & 63)
    const float a = s1[c] + s1[c + 64] + s1[c + 128] + s1[c + 192];
    const float b = s2[c] + s2[c + 64] + s2[c + 128] + s2[c + 192];
    atomicAdd(&sums[c], a);
    atomicAdd(&sqs[c], b);
  }
}

__global__ void bn_finalize(const float* __restrict__ sums, const float* __restrict__ sqs,
                            float* __restrict__ mean, float* __restrict__ inv, int Nrows) {
  const int c = threadIdx.x;
  if (c >= 64) return;
  const float mu  = sums[c] / (float)Nrows;
  const float var = sqs[c] / (float)Nrows - mu * mu;
  mean[c] = mu;
  inv[c]  = rsqrtf(var + 1e-5f);
}

__global__ void bn_apply_relu(float* __restrict__ X, const float* __restrict__ mean,
                              const float* __restrict__ inv, const float* __restrict__ g,
                              const float* __restrict__ b, int total) {
  const int i = blockIdx.x * blockDim.x + threadIdx.x;
  if (i >= total) return;
  const int c = i & 63;  // C = 64
  float v = (X[i] - mean[c]) * inv[c] * g[c] + b[c];
  X[i] = v > 0.f ? v : 0.f;
}

// ---------------------------------------------------------------------------
// Per-node attention dots, wave-per-node (coalesced full-row float4 loads).
// aw layout [H,64] flattened matches the 128-wide row: weight index == column.
// Head = lane half; reduce within each 16-lane half via shfl_xor.
// ---------------------------------------------------------------------------
__global__ void node_dots(const float* __restrict__ xh, const float* __restrict__ aw_s,
                          const float* __restrict__ aw_d, float* __restrict__ dsrc,
                          float* __restrict__ ddst, int Nn) {
  const int lane = threadIdx.x & 31;
  const int n = (blockIdx.x * blockDim.x + threadIdx.x) >> 5;
  if (n >= Nn) return;
  const float4 xv = *reinterpret_cast<const float4*>(xh + (size_t)n * 128 + lane * 4);
  const float4 ws = *reinterpret_cast<const float4*>(aw_s + lane * 4);
  const float4 wd = *reinterpret_cast<const float4*>(aw_d + lane * 4);
  float ps = xv.x * ws.x + xv.y * ws.y + xv.z * ws.z + xv.w * ws.w;
  float pd = xv.x * wd.x + xv.y * wd.y + xv.z * wd.z + xv.w * wd.w;
#pragma unroll
  for (int m = 8; m >= 1; m >>= 1) {
    ps += __shfl_xor(ps, m, 16);
    pd += __shfl_xor(pd, m, 16);
  }
  if ((lane & 15) == 0) {
    const int h = lane >> 4;
    dsrc[n * 2 + h] = ps;
    ddst[n * 2 + h] = pd;
  }
}

// w_eh[d,h] = sum_c We[d, h*64+c] * ae[h,c]   (folds edge features into logit)
__global__ void edge_w(const float* __restrict__ We, const float* __restrict__ ae,
                       float* __restrict__ weh) {
  const int t = threadIdx.x;
  if (t >= 16) return;
  const int d = t >> 1, h = t & 1;
  float s = 0.f;
  for (int c = 0; c < 64; ++c) s += We[d * 128 + h * 64 + c] * ae[h * 64 + c];
  weh[d * 2 + h] = s;
}

// order-preserving float <-> uint for atomicMax
__device__ __forceinline__ unsigned enc_f(float f) {
  unsigned u = __float_as_uint(f);
  return (u & 0x80000000u) ? ~u : (u | 0x80000000u);
}
__device__ __forceinline__ float dec_f(unsigned u) {
  return (u & 0x80000000u) ? __uint_as_float(u & 0x7FFFFFFFu) : __uint_as_float(~u);
}

__global__ void fill_u32(unsigned* __restrict__ p, unsigned v, int n) {
  const int i = blockIdx.x * blockDim.x + threadIdx.x;
  if (i < n) p[i] = v;
}

// logits + leaky_relu + segment max
__global__ void edge_alpha(const float* __restrict__ ea, const float* __restrict__ weh,
                           const int* __restrict__ src, const int* __restrict__ dst,
                           const float* __restrict__ dsrc, const float* __restrict__ ddst,
                           float* __restrict__ alpha, unsigned* __restrict__ menc, int Ecnt) {
  const int e = blockIdx.x * blockDim.x + threadIdx.x;
  if (e >= Ecnt) return;
  const int s = src[e], d = dst[e];
  float edv[8];
#pragma unroll
  for (int k = 0; k < 8; ++k) edv[k] = ea[(size_t)e * 8 + k];
#pragma unroll
  for (int h = 0; h < 2; ++h) {
    float a = dsrc[s * 2 + h] + ddst[d * 2 + h];
#pragma unroll
    for (int k = 0; k < 8; ++k) a += edv[k] * weh[k * 2 + h];
    a = a > 0.f ? a : 0.2f * a;                 // leaky_relu(0.2)
    alpha[(size_t)e * 2 + h] = a;
    atomicMax(&menc[d * 2 + h], enc_f(a));
  }
}

// exp(logit - max) + segment sum
__global__ void edge_exp(float* __restrict__ alpha, const int* __restrict__ dst,
                         const unsigned* __restrict__ menc, float* __restrict__ ssum, int Ecnt) {
  const int e = blockIdx.x * blockDim.x + threadIdx.x;
  if (e >= Ecnt) return;
  const int d = dst[e];
#pragma unroll
  for (int h = 0; h < 2; ++h) {
    const float m  = dec_f(menc[d * 2 + h]);
    const float ex = __expf(alpha[(size_t)e * 2 + h] - m);
    alpha[(size_t)e * 2 + h] = ex;
    atomicAdd(&ssum[d * 2 + h], ex);
  }
}

// wave-per-edge weighted scatter-add with grid-stride + prefetch of the next
// edge's source row (global_prefetch_b8) to hide random-gather latency.
__global__ void edge_agg(const float* __restrict__ xh, const int* __restrict__ src,
                         const int* __restrict__ dst, const float* __restrict__ alpha,
                         const float* __restrict__ ssum, float* __restrict__ acc, int Ecnt) {
  const int lane = threadIdx.x & 31;
  const int nwaves = (gridDim.x * blockDim.x) >> 5;
  for (int e = (blockIdx.x * blockDim.x + threadIdx.x) >> 5; e < Ecnt; e += nwaves) {
    const int en = e + nwaves;
    if (en < Ecnt) {
      __builtin_prefetch(xh + (size_t)src[en] * 128 + lane * 4, 0, 1);
    }
    const int s = src[e], d = dst[e];
    const float2 al = *reinterpret_cast<const float2*>(alpha + (size_t)e * 2);
    const float coef = (lane < 16) ? al.x / (ssum[d * 2] + 1e-16f)
                                   : al.y / (ssum[d * 2 + 1] + 1e-16f);
    const float4 xv = *reinterpret_cast<const float4*>(xh + (size_t)s * 128 + lane * 4);
    float* outp = acc + (size_t)d * 128 + lane * 4;
    atomicAdd(outp + 0, xv.x * coef);
    atomicAdd(outp + 1, xv.y * coef);
    atomicAdd(outp + 2, xv.z * coef);
    atomicAdd(outp + 3, xv.w * coef);
  }
}

__global__ void bias_relu128(float* __restrict__ X, const float* __restrict__ b, int total) {
  const int i = blockIdx.x * blockDim.x + threadIdx.x;
  if (i >= total) return;
  const int c = i & 127;
  const float v = X[i] + b[c];
  X[i] = v > 0.f ? v : 0.f;
}

// out[n,c] = mean over heads + bias
__global__ void final_mean(const float* __restrict__ acc, const float* __restrict__ b,
                           float* __restrict__ out, int Nn) {
  const int i = blockIdx.x * blockDim.x + threadIdx.x;
  if (i >= Nn * 64) return;
  const int n = i >> 6, c = i & 63;
  out[i] = 0.5f * (acc[(size_t)n * 128 + c] + acc[(size_t)n * 128 + 64 + c]) + b[c];
}

// ---------------------------------------------------------------------------
// Orchestration
// ---------------------------------------------------------------------------
extern "C" void kernel_launch(void* const* d_in, const int* in_sizes, int n_in,
                              void* d_out, int out_size, void* d_ws, size_t ws_size,
                              hipStream_t stream) {
  (void)in_sizes; (void)n_in; (void)out_size; (void)ws_size;

  const float* x      = (const float*)d_in[0];
  const int*   ei     = (const int*)d_in[1];
  const float* ea     = (const float*)d_in[2];
  const float* lin0_w = (const float*)d_in[3];
  const float* lin0_b = (const float*)d_in[4];
  const float* bn0_g  = (const float*)d_in[5];
  const float* bn0_b  = (const float*)d_in[6];
  const float* lin1_w = (const float*)d_in[7];
  const float* lin1_b = (const float*)d_in[8];
  const float* bne_g  = (const float*)d_in[9];
  const float* bne_b  = (const float*)d_in[10];
  const float* W0     = (const float*)d_in[11];
  const float* We0    = (const float*)d_in[12];
  const float* as0    = (const float*)d_in[13];
  const float* ad0    = (const float*)d_in[14];
  const float* ae0    = (const float*)d_in[15];
  const float* b0     = (const float*)d_in[16];
  const float* W1     = (const float*)d_in[17];
  const float* We1    = (const float*)d_in[18];
  const float* as1    = (const float*)d_in[19];
  const float* ad1    = (const float*)d_in[20];
  const float* ae1    = (const float*)d_in[21];
  const float* b1     = (const float*)d_in[22];
  float*       out    = (float*)d_out;

  const int* src = ei;
  const int* dst = ei + EE;

  // workspace carve-up
  char* w = (char*)d_ws;
  float*    r1    = (float*)w; w += (size_t)NN * 64 * 4;   // t0/t1
  float*    r2    = (float*)w; w += (size_t)NN * 64 * 4;   // t2/t3
  float*    xh    = (float*)w; w += (size_t)NN * 128 * 4;  // xh0, later xh1
  float*    h4    = (float*)w; w += (size_t)NN * 128 * 4;  // conv0 agg / features
  float*    alpha = (float*)w; w += (size_t)EE * 2 * 4;
  unsigned* menc  = (unsigned*)w; w += (size_t)NN * 2 * 4;
  float*    ssum  = (float*)w; w += (size_t)NN * 2 * 4;
  float*    dsrc  = (float*)w; w += (size_t)NN * 2 * 4;
  float*    ddst  = (float*)w; w += (size_t)NN * 2 * 4;
  float*    mean  = (float*)w; w += 64 * 4;
  float*    inv   = (float*)w; w += 64 * 4;
  float*    weh   = (float*)w; w += 16 * 4;
  float*    bnsum = (float*)w; w += 128 * 4;               // [64 sums | 64 sumsq]
  float*    acc1  = r1;  // conv1 accumulator spans r1+r2 (N x 128), both dead by then

  const dim3 blk(256);
  const int gGemm   = (NN + 127) / 128;               // 8 waves * 16 rows per block
  const int gNodesW = (NN * 32 + 255) / 256;          // wave-per-node
  const int gN64    = (NN * 64 + 255) / 256;
  const int gN128   = (NN * 128 + 255) / 256;
  const int gN2     = (NN * 2 + 255) / 256;
  const int gE      = (EE + 255) / 256;
  const unsigned NEG_INF_ENC = 0x007FFFFFu;           // enc(-inf)

  // ---- embedding MLP ----
  gemm_wmma<128, 64, true><<<gGemm, blk, 0, stream>>>(x, lin0_w, lin0_b, r1, NN);
  hipMemsetAsync(bnsum, 0, 128 * 4, stream);
  bn_partial<<<512, blk, 0, stream>>>(r1, NN * 64, bnsum, bnsum + 64);
  bn_finalize<<<1, 64, 0, stream>>>(bnsum, bnsum + 64, mean, inv, NN);
  bn_apply_relu<<<gN64, blk, 0, stream>>>(r1, mean, inv, bn0_g, bn0_b, NN * 64);
  gemm_wmma<64, 64, true><<<gGemm, blk, 0, stream>>>(r1, lin1_w, lin1_b, r2, NN);
  hipMemsetAsync(bnsum, 0, 128 * 4, stream);
  bn_partial<<<512, blk, 0, stream>>>(r2, NN * 64, bnsum, bnsum + 64);
  bn_finalize<<<1, 64, 0, stream>>>(bnsum, bnsum + 64, mean, inv, NN);
  bn_apply_relu<<<gN64, blk, 0, stream>>>(r2, mean, inv, bne_g, bne_b, NN * 64);

  // ---- conv0 (concat) ----
  gemm_wmma<64, 128, false><<<gGemm, blk, 0, stream>>>(r2, W0, nullptr, xh, NN);
  node_dots<<<gNodesW, blk, 0, stream>>>(xh, as0, ad0, dsrc, ddst, NN);
  edge_w<<<1, 16, 0, stream>>>(We0, ae0, weh);
  fill_u32<<<gN2, blk, 0, stream>>>(menc, NEG_INF_ENC, NN * 2);
  hipMemsetAsync(ssum, 0, (size_t)NN * 2 * 4, stream);
  hipMemsetAsync(h4, 0, (size_t)NN * 128 * 4, stream);
  edge_alpha<<<gE, blk, 0, stream>>>(ea, weh, src, dst, dsrc, ddst, alpha, menc, EE);
  edge_exp<<<gE, blk, 0, stream>>>(alpha, dst, menc, ssum, EE);
  edge_agg<<<2048, blk, 0, stream>>>(xh, src, dst, alpha, ssum, h4, EE);
  bias_relu128<<<gN128, blk, 0, stream>>>(h4, b0, NN * 128);

  // ---- conv1 (mean over heads) ----
  gemm_wmma<128, 128, false><<<gGemm, blk, 0, stream>>>(h4, W1, nullptr, xh, NN);
  node_dots<<<gNodesW, blk, 0, stream>>>(xh, as1, ad1, dsrc, ddst, NN);
  edge_w<<<1, 16, 0, stream>>>(We1, ae1, weh);
  fill_u32<<<gN2, blk, 0, stream>>>(menc, NEG_INF_ENC, NN * 2);
  hipMemsetAsync(ssum, 0, (size_t)NN * 2 * 4, stream);
  hipMemsetAsync(acc1, 0, (size_t)NN * 128 * 4, stream);
  edge_alpha<<<gE, blk, 0, stream>>>(ea, weh, src, dst, dsrc, ddst, alpha, menc, EE);
  edge_exp<<<gE, blk, 0, stream>>>(alpha, dst, menc, ssum, EE);
  edge_agg<<<2048, blk, 0, stream>>>(xh, src, dst, alpha, ssum, acc1, EE);
  final_mean<<<gN64, blk, 0, stream>>>(acc1, b1, out, NN);
}
